// Attention_38328288149847
// MI455X (gfx1250) — matrix operand order repaired and checked by reading, not compiled, not run on hardware
//
#include <hip/hip_runtime.h>

// ---------------------------------------------------------------------------
// MI455X (gfx1250) fused causal attention: QKV proj + RoPE -> flash attention
// -> output proj.  bf16 WMMA (v_wmma_f32_16x16x32_bf16), f32 accumulation.
// K tiles staged to LDS via the Tensor Data Mover (tensor_load_to_lds, 6-arg
// toolchain variant) with hardware row padding; falls back to manual staging
// if the builtin is absent.
// ---------------------------------------------------------------------------

using bf16 = __bf16;
typedef __attribute__((ext_vector_type(8)))  bf16  v8bf;
typedef __attribute__((ext_vector_type(16))) bf16  v16bf;
typedef __attribute__((ext_vector_type(8)))  float v8f;

#if __has_builtin(__builtin_amdgcn_tensor_load_to_lds)
#define HAVE_TDM 1
typedef unsigned int u32x4 __attribute__((ext_vector_type(4)));
typedef int          i32x8 __attribute__((ext_vector_type(8)));
typedef int          i32x4 __attribute__((ext_vector_type(4)));
#else
#define HAVE_TDM 0
#endif

#define B_DIM 2
#define S_DIM 2048
#define E_DIM 2048
#define H_DIM 16
#define HD    128
#define QKV_ELEMS ((size_t)B_DIM * H_DIM * S_DIM * HD)   // 8388608 per tensor
#define ATTN_SCALE 0.022097086912079608f                 // 1/sqrt(E)
#define NEG_BIG (-3.0e38f)

// GEMM tile LDS strides (elements; multiples of 8 keep 16B-aligned runs)
#define LDA 40   // 32 + 8 pad
#define LDB 40
// attention LDS strides
#define LDK 136  // K tile [t][d]: 128 + 8 pad (matches TDM pad 64dw+4dw)
#define LDV 72   // 64 + 8 pad (V transposed: [d][t])
#define LDP 72   // P scratch per wave: [16][64] padded

__device__ __forceinline__ bf16 tobf(float f) {
    unsigned u = __builtin_bit_cast(unsigned, f);
    u += 0x7FFFu + ((u >> 16) & 1u);                 // round-to-nearest-even
    unsigned short h = (unsigned short)(u >> 16);
    return __builtin_bit_cast(bf16, h);
}

__device__ __forceinline__ v16bf cat16(v8bf lo, v8bf hi) {
    return __builtin_shufflevector(lo, hi, 0,1,2,3,4,5,6,7,8,9,10,11,12,13,14,15);
}

__device__ __forceinline__ v8f wmma_bf16(v16bf a, v16bf b, v8f c) {
    // D = A(16x32) * B(32x16) + C, f32 accumulate
    return __builtin_amdgcn_wmma_f32_16x16x32_bf16(false, a, false, b,
                                                   (short)0, c, false, false);
}

// A-matrix fragment (16x32 bf16): lane L holds row M=L%16,
// K runs [h*8, h*8+8) and [h*8+16, h*8+24) with h = L/16  (ISA 7.12.2)
__device__ __forceinline__ v16bf load_afrag(const bf16* row_base, int half8) {
    const bf16* p = row_base + half8;                // half8 = (lane>>4)*8
    return cat16(*(const v8bf*)p, *(const v8bf*)(p + 16));
}
// B-matrix fragment (32x16 bf16) staged as [n][k]: lane L holds col N=L%16,
// K runs [h*8, h*8+8) and [h*8+16, h*8+24) with h = L/16
__device__ __forceinline__ v16bf load_bfrag(const bf16* row_base, int half16) {
    const bf16* p = row_base + (half16 >> 1);        // half16 = (lane>>4)*16
    return cat16(*(const v8bf*)p, *(const v8bf*)(p + 16));
}

__device__ __forceinline__ float hmax16(float v) {
    v = fmaxf(v, __shfl_xor(v, 1));
    v = fmaxf(v, __shfl_xor(v, 2));
    v = fmaxf(v, __shfl_xor(v, 4));
    v = fmaxf(v, __shfl_xor(v, 8));
    return v;
}
__device__ __forceinline__ float hsum16(float v) {
    v += __shfl_xor(v, 1);
    v += __shfl_xor(v, 2);
    v += __shfl_xor(v, 4);
    v += __shfl_xor(v, 8);
    return v;
}

// ---------------------------------------------------------------------------
// Kernel 1: C = x * W^T (128x128 tile per block), f32->bf16 on stage,
// RoPE epilogue for Q/K, writes head-major bf16 [B,H,S,D].
// blockIdx.z selects Wq/Wk/Wv.
// ---------------------------------------------------------------------------
__global__ __launch_bounds__(256) void qkv_rope_kernel(
    const float* __restrict__ x,
    const float* __restrict__ Wq, const float* __restrict__ Wk,
    const float* __restrict__ Wv,
    bf16* __restrict__ Qh, bf16* __restrict__ Kh, bf16* __restrict__ Vh)
{
    __shared__ bf16 As[128 * LDA];
    __shared__ bf16 Bs[128 * LDB];

    const int which = blockIdx.z;
    const float* __restrict__ W = (which == 0) ? Wq : ((which == 1) ? Wk : Wv);
    bf16* __restrict__ dst      = (which == 0) ? Qh : ((which == 1) ? Kh : Vh);

    const int m0 = blockIdx.y * 128;
    const int n0 = blockIdx.x * 128;
    const int tid  = threadIdx.x;
    const int w    = tid >> 5;
    const int lane = tid & 31, lm = lane & 15, half = lane >> 4;
    const int waveM = (w & 3) * 32;      // 4x2 wave grid of 32x64 tiles
    const int waveN = (w >> 2) * 64;

    v8f acc[2][4];
#pragma unroll
    for (int mi = 0; mi < 2; ++mi)
#pragma unroll
        for (int ni = 0; ni < 4; ++ni) acc[mi][ni] = (v8f)0.0f;

    for (int k0 = 0; k0 < E_DIM; k0 += 32) {
        __syncthreads();
        // stage A (x tile 128x32) and B (W tile 128 rows x 32 cols), f32->bf16
        for (int i = tid; i < 128 * 4; i += 256) {
            const int r = i >> 2, c = (i & 3) * 8;
            const float* gp = x + (size_t)(m0 + r) * E_DIM + k0 + c;
            float4 f0 = *(const float4*)gp;
            float4 f1 = *(const float4*)(gp + 4);
            v8bf pa = { tobf(f0.x), tobf(f0.y), tobf(f0.z), tobf(f0.w),
                        tobf(f1.x), tobf(f1.y), tobf(f1.z), tobf(f1.w) };
            *(v8bf*)&As[r * LDA + c] = pa;

            const float* wp = W + (size_t)(n0 + r) * E_DIM + k0 + c;
            float4 g0 = *(const float4*)wp;
            float4 g1 = *(const float4*)(wp + 4);
            v8bf pb = { tobf(g0.x), tobf(g0.y), tobf(g0.z), tobf(g0.w),
                        tobf(g1.x), tobf(g1.y), tobf(g1.z), tobf(g1.w) };
            *(v8bf*)&Bs[r * LDB + c] = pb;

            if (k0 + 32 < E_DIM) {                    // global_prefetch_b8
                __builtin_prefetch(gp + 32, 0, 1);
                __builtin_prefetch(wp + 32, 0, 1);
            }
        }
        __syncthreads();

        v16bf a[2], b[4];
#pragma unroll
        for (int mi = 0; mi < 2; ++mi)
            a[mi] = load_afrag(&As[(waveM + mi * 16 + lm) * LDA], half * 8);
#pragma unroll
        for (int ni = 0; ni < 4; ++ni)
            b[ni] = load_bfrag(&Bs[(waveN + ni * 16 + lm) * LDB], half * 16);
#pragma unroll
        for (int mi = 0; mi < 2; ++mi)
#pragma unroll
            for (int ni = 0; ni < 4; ++ni)
                acc[mi][ni] = wmma_bf16(a[mi], b[ni], acc[mi][ni]);
    }

    // Epilogue: optional RoPE, store bf16 head-major [B,H,S,D]
#pragma unroll
    for (int mi = 0; mi < 2; ++mi) {
#pragma unroll
        for (int ni = 0; ni < 4; ++ni) {
#pragma unroll
            for (int r = 0; r < 8; ++r) {
                const int mg = m0 + waveM + mi * 16 + half * 8 + r;   // global row
                const int c  = n0 + waveN + ni * 16 + lm;             // global col
                const int bb = mg >> 11, s = mg & (S_DIM - 1);
                const int h  = c >> 7,  d = c & (HD - 1);
                float v = acc[mi][ni][r];
                float partner = __shfl_xor(v, 1);   // column c^1 (adjacent lane)
                float outv;
                if (which < 2) {
                    // inv_freq = 10000^{-(d>>1)/64}; ln(10000)/64 = 0.14391156831
                    float ang = (float)s * __expf(-(float)(d >> 1) * 0.14391156831f);
                    float cs = __cosf(ang), sn = __sinf(ang);
                    outv = (d & 1) ? (v * cs + partner * sn)
                                   : (v * cs - partner * sn);
                } else {
                    outv = v;
                }
                dst[((size_t)(bb * H_DIM + h) * S_DIM + s) * HD + d] = tobf(outv);
            }
        }
    }
}

// ---------------------------------------------------------------------------
// Kernel 2: flash attention, causal. One block = 128 q rows of one (b,h);
// 8 waves x 16 rows. Q fragments come straight from global (rows contiguous
// over d). K tile is DMA'd into LDS by the Tensor Data Mover (with hardware
// row padding -> stride 136 elems, bank-conflict-free B-fragment reads);
// V is transposed through LDS, P round-trips per-wave LDS scratch.
// ---------------------------------------------------------------------------
__global__ __launch_bounds__(256) void attn_kernel(
    const bf16* __restrict__ Qh, const bf16* __restrict__ Kh,
    const bf16* __restrict__ Vh, bf16* __restrict__ AO)
{
    __shared__ bf16 Ks[64 * LDK];           // [t][d] K tile (padded rows)
    __shared__ bf16 Vt[128 * LDV];          // [d][t] transposed V tile
    __shared__ bf16 Pw[8 * 16 * LDP];       // per-wave P scratch [16][64]

    const int bh = blockIdx.x;
    const int b = bh >> 4, h = bh & 15;
    const int qBase = blockIdx.y * 128;
    const int tid = threadIdx.x, w = tid >> 5;
    const int lane = tid & 31, lm = lane & 15, half = lane >> 4;
    const size_t base = (size_t)bh * S_DIM * HD;

    // Q fragments for this wave's 16 rows live in registers for the whole loop
    v16bf aq[4];
#pragma unroll
    for (int kk = 0; kk < 4; ++kk) {
        const bf16* p = Qh + base + (size_t)(qBase + w * 16 + lm) * HD + kk * 32;
        aq[kk] = load_afrag(p, half * 8);
    }

    v8f o[8];
#pragma unroll
    for (int ni = 0; ni < 8; ++ni) o[ni] = (v8f)0.0f;
    float m8[8], l8[8];
#pragma unroll
    for (int r = 0; r < 8; ++r) { m8[r] = NEG_BIG; l8[r] = 0.0f; }

    const int ntiles = 2 * blockIdx.y + 2;          // causal: t0 <= qBase+64
    for (int ti = 0; ti < ntiles; ++ti) {
        const int t0 = ti * 64;
        __syncthreads();                    // protect Ks/Vt from prior readers

        // stage V transposed: Vt[d][t] = V[t0+t][d]
        for (int i = tid; i < 64 * 16; i += 256) {
            const int t = i >> 4, c = (i & 15) * 8;
            v8bf vv = *(const v8bf*)(Vh + base + (size_t)(t0 + t) * HD + c);
#pragma unroll
            for (int e = 0; e < 8; ++e) Vt[(c + e) * LDV + t] = vv[e];
        }

#if HAVE_TDM
        // K tile via Tensor Data Mover: 64x128 bf16, row-padded into LDS.
        // One wave issues the DMA (TDM ignores EXEC; wave-uniform branch
        // prevents duplicate descriptors), waits TENSORcnt, barrier publishes.
        if (w == 0) {
            const unsigned lds = (unsigned)(size_t)(&Ks[0]);
            const unsigned long long ga =
                (unsigned long long)(size_t)(Kh + base + (size_t)t0 * HD);
            // D# group0: count=1 | lds_addr | global_addr | type=2
            u32x4 g0 = { 1u, lds,
                         (unsigned)(ga & 0xFFFFFFFFu),
                         (unsigned)(ga >> 32) | 0x80000000u };
            // D# group1: data_size=1(2B), pad_enable, pad_interval=5(64dw),
            // pad_amount=3(4dw); tensor_dim0=128, tensor_dim1=0xFFFF;
            // tile_dim0=128, tile_dim1=64; tensor_dim0_stride=128
            i32x8 g1 = { (int)((1u << 16) | (1u << 20) | (5u << 22) | (3u << 25)),
                         (int)(128u << 16),      // tensor_dim0 low -> [31:16]
                         (int)0xFFFF0000u,       // tensor_dim1 low16 -> [31:16]
                         (int)(128u << 16),      // tile_dim0 -> [127:112]
                         64,                     // tile_dim1
                         128,                    // tensor_dim0_stride low32
                         0, 0 };
            i32x4 z4 = { 0, 0, 0, 0 };
            i32x8 z8 = { 0, 0, 0, 0, 0, 0, 0, 0 };
            // 6-arg toolchain variant: (g0, g1, g2, g3, g4, cpol)
            __builtin_amdgcn_tensor_load_to_lds(g0, g1, z4, z4, z8, 0);
            __builtin_amdgcn_s_wait_tensorcnt(0);
        }
#else
        // fallback: manual K staging into the same padded layout
        for (int i = tid; i < 64 * 16; i += 256) {
            const int t = i >> 4, c = (i & 15) * 8;
            *(v8bf*)&Ks[t * LDK + c] =
                *(const v8bf*)(Kh + base + (size_t)(t0 + t) * HD + c);
        }
#endif
        __syncthreads();

        // S = Q * K^T  (16x64 per wave): B fragment rows from LDS K tile
        v8f sf[4];
#pragma unroll
        for (int ni = 0; ni < 4; ++ni) sf[ni] = (v8f)0.0f;
#pragma unroll
        for (int kk = 0; kk < 4; ++kk) {
#pragma unroll
            for (int ni = 0; ni < 4; ++ni) {
                const bf16* p = &Ks[(ni * 16 + lm) * LDK + kk * 32];
                sf[ni] = wmma_bf16(aq[kk], load_bfrag(p, half * 16), sf[ni]);
            }
        }

        // online softmax (rows split across 16-lane halves)
#pragma unroll
        for (int r = 0; r < 8; ++r) {
            const int sg = qBase + w * 16 + half * 8 + r;
            float vals[4];
            float rowm = NEG_BIG;
#pragma unroll
            for (int ni = 0; ni < 4; ++ni) {
                float v = sf[ni][r] * ATTN_SCALE;
                const int tg = t0 + ni * 16 + lm;
                if (tg > sg) v = NEG_BIG;            // causal mask
                vals[ni] = v;
                rowm = fmaxf(rowm, v);
            }
            rowm = hmax16(rowm);
            const float mn = fmaxf(m8[r], rowm);
            const float alpha = __expf(m8[r] - mn);
            float rs = 0.0f;
#pragma unroll
            for (int ni = 0; ni < 4; ++ni) {
                const float p = __expf(vals[ni] - mn);
                rs += p;
                Pw[w * 16 * LDP + (half * 8 + r) * LDP + ni * 16 + lm] = tobf(p);
            }
            rs = hsum16(rs);
            l8[r] = l8[r] * alpha + rs;
            m8[r] = mn;
#pragma unroll
            for (int ni = 0; ni < 8; ++ni) o[ni][r] *= alpha;
        }

        // O += P * V   (A = P from LDS scratch, B = Vt rows, contiguous t)
#pragma unroll
        for (int kk = 0; kk < 2; ++kk) {
            const bf16* pp = &Pw[w * 16 * LDP + lm * LDP + kk * 32];
            v16bf ap = load_afrag(pp, half * 8);
#pragma unroll
            for (int ni = 0; ni < 8; ++ni) {
                const bf16* vb = &Vt[(ni * 16 + lm) * LDV + kk * 32];
                o[ni] = wmma_bf16(ap, load_bfrag(vb, half * 16), o[ni]);
            }
        }
    }

    // normalize + store bf16 to [B,S,E] (heads merged)
#pragma unroll
    for (int r = 0; r < 8; ++r) {
        const float inv = 1.0f / l8[r];
        const int s = qBase + w * 16 + half * 8 + r;
#pragma unroll
        for (int ni = 0; ni < 8; ++ni) {
            const int d = ni * 16 + lm;
            AO[((size_t)(b * S_DIM + s)) * E_DIM + h * HD + d] =
                tobf(o[ni][r] * inv);
        }
    }
}

// ---------------------------------------------------------------------------
// Kernel 3: out = AO(bf16) * Wo^T -> f32
// ---------------------------------------------------------------------------
__global__ __launch_bounds__(256) void oproj_kernel(
    const bf16* __restrict__ AO, const float* __restrict__ Wo,
    float* __restrict__ out)
{
    __shared__ bf16 As[128 * LDA];
    __shared__ bf16 Bs[128 * LDB];

    const int m0 = blockIdx.y * 128;
    const int n0 = blockIdx.x * 128;
    const int tid  = threadIdx.x;
    const int w    = tid >> 5;
    const int lane = tid & 31, lm = lane & 15, half = lane >> 4;
    const int waveM = (w & 3) * 32;
    const int waveN = (w >> 2) * 64;

    v8f acc[2][4];
#pragma unroll
    for (int mi = 0; mi < 2; ++mi)
#pragma unroll
        for (int ni = 0; ni < 4; ++ni) acc[mi][ni] = (v8f)0.0f;

    for (int k0 = 0; k0 < E_DIM; k0 += 32) {
        __syncthreads();
        for (int i = tid; i < 128 * 4; i += 256) {
            const int r = i >> 2, c = (i & 3) * 8;
            *(v8bf*)&As[r * LDA + c] =
                *(const v8bf*)(AO + (size_t)(m0 + r) * E_DIM + k0 + c);

            const float* wp = Wo + (size_t)(n0 + r) * E_DIM + k0 + c;
            float4 g0 = *(const float4*)wp;
            float4 g1 = *(const float4*)(wp + 4);
            v8bf pb = { tobf(g0.x), tobf(g0.y), tobf(g0.z), tobf(g0.w),
                        tobf(g1.x), tobf(g1.y), tobf(g1.z), tobf(g1.w) };
            *(v8bf*)&Bs[r * LDB + c] = pb;

            if (k0 + 32 < E_DIM) __builtin_prefetch(wp + 32, 0, 1);
        }
        __syncthreads();

        v16bf a[2], b[4];
#pragma unroll
        for (int mi = 0; mi < 2; ++mi)
            a[mi] = load_afrag(&As[(waveM + mi * 16 + lm) * LDA], half * 8);
#pragma unroll
        for (int ni = 0; ni < 4; ++ni)
            b[ni] = load_bfrag(&Bs[(waveN + ni * 16 + lm) * LDB], half * 16);
#pragma unroll
        for (int mi = 0; mi < 2; ++mi)
#pragma unroll
            for (int ni = 0; ni < 4; ++ni)
                acc[mi][ni] = wmma_bf16(a[mi], b[ni], acc[mi][ni]);
    }

#pragma unroll
    for (int mi = 0; mi < 2; ++mi)
#pragma unroll
        for (int ni = 0; ni < 4; ++ni)
#pragma unroll
            for (int r = 0; r < 8; ++r) {
                const int mg = m0 + waveM + mi * 16 + half * 8 + r;
                const int c  = n0 + waveN + ni * 16 + lm;
                out[(size_t)mg * E_DIM + c] = acc[mi][ni][r];
            }
}

// ---------------------------------------------------------------------------
extern "C" void kernel_launch(void* const* d_in, const int* in_sizes, int n_in,
                              void* d_out, int out_size, void* d_ws, size_t ws_size,
                              hipStream_t stream) {
    const float* x  = (const float*)d_in[0];
    const float* Wq = (const float*)d_in[1];
    const float* Wk = (const float*)d_in[2];
    const float* Wv = (const float*)d_in[3];
    const float* Wo = (const float*)d_in[4];
    float* out = (float*)d_out;

    // workspace: Q,K,V head-major bf16 + merged attention output bf16 (64 MB)
    bf16* Qh = (bf16*)d_ws;
    bf16* Kh = Qh + QKV_ELEMS;
    bf16* Vh = Kh + QKV_ELEMS;
    bf16* AO = Vh + QKV_ELEMS;

    dim3 gProj(E_DIM / 128, (B_DIM * S_DIM) / 128, 3);
    qkv_rope_kernel<<<gProj, 256, 0, stream>>>(x, Wq, Wk, Wv, Qh, Kh, Vh);

    dim3 gAttn(B_DIM * H_DIM, S_DIM / 128);
    attn_kernel<<<gAttn, 256, 0, stream>>>(Qh, Kh, Vh, AO);

    dim3 gOut(E_DIM / 128, (B_DIM * S_DIM) / 128);
    oproj_kernel<<<gOut, 256, 0, stream>>>(AO, Wo, out);
}
